// AntiviralGNN_15247133901100
// MI455X (gfx1250) — matrix operand order repaired and checked by reading, not compiled, not run on hardware
//
#include <hip/hip_runtime.h>
#include <hip/hip_bf16.h>
#include <cstdint>
#include <cstddef>

typedef __attribute__((ext_vector_type(16))) _Float16 v16h;
typedef __attribute__((ext_vector_type(8)))  _Float16 v8h;
typedef __attribute__((ext_vector_type(8)))  float    v8f;

// ---------------- small helpers ----------------

__device__ __forceinline__ void atomicMaxF(float* addr, float v) {
  // standard sign-split trick: works for mixed-sign values with -inf init
  if (v >= 0.0f) atomicMax((int*)addr, __float_as_int(v));
  else           atomicMin((unsigned int*)addr, __float_as_uint(v));
}

__global__ void k_fill(float* __restrict__ p, float v, int n) {
  int i = blockIdx.x * blockDim.x + threadIdx.x;
  if (i < n) p[i] = v;
}

// X [n,K] f32 row-major -> Xh [n,Kpad] f16 (zero padded K..Kpad)
__global__ void k_convert_pad_half(const float* __restrict__ X, _Float16* __restrict__ Xh,
                                   int n, int K, int Kpad) {
  int i = blockIdx.x * blockDim.x + threadIdx.x;
  if (i >= n * Kpad) return;
  int row = i / Kpad, k = i - row * Kpad;
  Xh[i] = (k < K) ? (_Float16)X[(size_t)row * K + k] : (_Float16)0.0f;
}

// W [K,M] f32 row-major -> WT [M,Kpad] f16 (transposed, zero padded)
__global__ void k_transpose_half(const float* __restrict__ W, _Float16* __restrict__ WT,
                                 int K, int M, int Kpad) {
  int i = blockIdx.x * blockDim.x + threadIdx.x;
  if (i >= M * Kpad) return;
  int m = i / Kpad, k = i - m * Kpad;
  WT[i] = (k < K) ? (_Float16)W[(size_t)k * M + m] : (_Float16)0.0f;
}

// ---------------- WMMA GEMM: Y[n,M] = Ah[n,Kpad] * Bt[M,Kpad]^T ----------------
// One wave32 computes one 16x16 tile via v_wmma_f32_16x16x32_f16.
// A fragment (16x32 f16): lane&15 = row; lanes 0-15 hold K 0..7,16..23; lanes 16-31 K 8..15,24..31.
// B fragment (32x16 f16): lane&15 = col; lanes 0-15 hold K 0..7,16..23; lanes 16-31 K 8..15,24..31.
// C/D (16x16 f32): VGPR v -> row v (lanes 0-15) / v+8 (lanes 16-31), col = lane&15.
__global__ void k_wmma_gemm(const _Float16* __restrict__ Ah, const _Float16* __restrict__ Bt,
                            float* __restrict__ Y, int nrows, int M, int Kpad) {
  int wid  = (blockIdx.x * blockDim.x + threadIdx.x) >> 5;
  int lane = threadIdx.x & 31;
  int tiles_n = M >> 4;
  int total = (nrows >> 4) * tiles_n;
  if (wid >= total) return;                 // wave-uniform: EXEC stays all-ones
  int tm = wid / tiles_n, tn = wid - tm * tiles_n;
  int l15 = lane & 15;
  int hi  = lane >> 4;
  const _Float16* Ap = Ah + (size_t)(tm * 16 + l15) * Kpad + hi * 8;
  const _Float16* Bp = Bt + (size_t)(tn * 16 + l15) * Kpad + hi * 8;
  v8f acc = {};
  for (int k0 = 0; k0 < Kpad; k0 += 32) {
    union { v16h v; v8h h[2]; } a, b;
    a.h[0] = *(const v8h*)(Ap + k0);
    a.h[1] = *(const v8h*)(Ap + k0 + 16);
    b.h[0] = *(const v8h*)(Bp + k0);
    b.h[1] = *(const v8h*)(Bp + k0 + 16);
    acc = __builtin_amdgcn_wmma_f32_16x16x32_f16(false, a.v, false, b.v,
                                                 (short)0, acc, false, false);
  }
  float* Yp = Y + (size_t)(tm * 16 + hi * 8) * M + tn * 16 + l15;
#pragma unroll
  for (int v = 0; v < 8; ++v) Yp[(size_t)v * M] = acc[v];
}

// ---------------- GATv2 edge kernels (one wave32 per edge) ----------------

// e[eh] = sum_c att[h,c] * leaky_relu(xl[src,h,c] + xr[dst,h,c]); segment max into m[dst,h]
__global__ void k_edge_logits(const float* __restrict__ xl, const float* __restrict__ xr,
                              const float* __restrict__ att,
                              const int* __restrict__ srcA, const int* __restrict__ dstA,
                              float* __restrict__ m, float* __restrict__ ew,
                              int E, int N, int H) {
  int wid  = (blockIdx.x * blockDim.x + threadIdx.x) >> 5;
  int lane = threadIdx.x & 31;
  int EL = E + N;
  if (wid >= EL) return;
  int s = (wid < E) ? srcA[wid] : (wid - E);   // self-loops appended
  int d = (wid < E) ? dstA[wid] : (wid - E);
  int HID = H * 64;
  const float* pl = xl + (size_t)s * HID;
  const float* pr = xr + (size_t)d * HID;
  for (int h = 0; h < H; ++h) {
    int c0 = h * 64 + lane;
    float u0 = pl[c0]      + pr[c0];
    float u1 = pl[c0 + 32] + pr[c0 + 32];
    u0 = (u0 > 0.0f) ? u0 : 0.2f * u0;
    u1 = (u1 > 0.0f) ? u1 : 0.2f * u1;
    float part = att[c0] * u0 + att[c0 + 32] * u1;
#pragma unroll
    for (int off = 16; off > 0; off >>= 1) part += __shfl_xor(part, off, 32);
    if (lane == 0) {
      ew[(size_t)wid * H + h] = part;
      atomicMaxF(&m[(size_t)d * H + h], part);
    }
  }
}

// p = exp(e - m[dst]); denom[dst] += p  (thread per edge*head)
__global__ void k_edge_exp(const int* __restrict__ dstA,
                           const float* __restrict__ m, float* __restrict__ ew,
                           float* __restrict__ denom, int E, int N, int H) {
  int i = blockIdx.x * blockDim.x + threadIdx.x;
  int EL = E + N;
  if (i >= EL * H) return;
  int e = i / H, h = i - e * H;
  int d = (e < E) ? dstA[e] : (e - E);
  float p = expf(ew[i] - m[(size_t)d * H + h]);
  ew[i] = p;
  atomicAdd(&denom[(size_t)d * H + h], p);
}

// agg[dst, c] += (p / denom[dst,h]) * xl[src, c]   (wave per edge, lane-strided channels)
__global__ void k_edge_aggregate(const float* __restrict__ xl, const float* __restrict__ ew,
                                 const float* __restrict__ denom,
                                 const int* __restrict__ srcA, const int* __restrict__ dstA,
                                 float* __restrict__ agg, int E, int N, int H) {
  int wid  = (blockIdx.x * blockDim.x + threadIdx.x) >> 5;
  int lane = threadIdx.x & 31;
  int EL = E + N;
  if (wid >= EL) return;
  int s = (wid < E) ? srcA[wid] : (wid - E);
  int d = (wid < E) ? dstA[wid] : (wid - E);
  int HID = H * 64;
  const float* pl = xl + (size_t)s * HID;
  float* po = agg + (size_t)d * HID;
  for (int c = lane; c < HID; c += 32) {
    int h = c >> 6;
    float alpha = ew[(size_t)wid * H + h] / (denom[(size_t)d * H + h] + 1e-16f);
    atomicAdd(&po[c], alpha * pl[c]);
  }
}

// ---------------- bias + BN ----------------

// add bias in place, accumulate per-channel sum/sumsq (blockDim.x == HID)
__global__ void k_bias_bnstats(float* __restrict__ y, const float* __restrict__ bias,
                               float* __restrict__ bsum, float* __restrict__ bsumsq,
                               int n, int HID, int rows_per_block) {
  int ch = threadIdx.x;
  int r0 = blockIdx.x * rows_per_block;
  float b = bias[ch];
  float s = 0.0f, s2 = 0.0f;
  for (int r = 0; r < rows_per_block; ++r) {
    int node = r0 + r;
    if (node >= n) break;                      // uniform across block
    size_t idx = (size_t)node * HID + ch;
    float v = y[idx] + b;
    y[idx] = v;
    s += v; s2 += v * v;
  }
  atomicAdd(&bsum[ch], s);
  atomicAdd(&bsumsq[ch], s2);
}

__global__ void k_bn_apply(const float* __restrict__ y,
                           const float* __restrict__ bsum, const float* __restrict__ bsumsq,
                           const float* __restrict__ g, const float* __restrict__ be,
                           float* __restrict__ fout, _Float16* __restrict__ hout,
                           int n, int HID, int do_elu) {
  int i = blockIdx.x * blockDim.x + threadIdx.x;
  if (i >= n * HID) return;
  int ch = i % HID;
  float invn = 1.0f / (float)n;
  float mu  = bsum[ch] * invn;
  float var = bsumsq[ch] * invn - mu * mu;
  float v = (y[i] - mu) * rsqrtf(var + 1e-5f) * g[ch] + be[ch];
  if (do_elu) v = (v > 0.0f) ? v : (expf(v) - 1.0f);
  if (fout) fout[i] = v;
  if (hout) hout[i] = (_Float16)v;
}

// ---------------- pooling + classifier ----------------

__global__ void k_pool_scatter(const float* __restrict__ h3, const int* __restrict__ batch,
                               float* __restrict__ gsum, float* __restrict__ gmax,
                               float* __restrict__ gcnt, int n) {
  int i = blockIdx.x * blockDim.x + threadIdx.x;
  if (i >= n * 64) return;
  int node = i >> 6, c = i & 63;
  int g = batch[node];
  float v = h3[i];
  atomicAdd(&gsum[g * 64 + c], v);
  atomicMaxF(&gmax[g * 64 + c], v);
  if (c == 0) atomicAdd(&gcnt[g], 1.0f);
}

__global__ void k_pool_finalize(const float* __restrict__ gsum, const float* __restrict__ gmax,
                                const float* __restrict__ gcnt, float* __restrict__ pooled,
                                int G) {
  int i = blockIdx.x * blockDim.x + threadIdx.x;
  if (i >= G * 128) return;
  int g = i >> 7, c = i & 127;
  pooled[i] = (c < 64) ? gsum[g * 64 + c] / fmaxf(gcnt[g], 1.0f)
                       : gmax[g * 64 + (c - 64)];
}

__global__ void k_fc1(const float* __restrict__ pooled, const float* __restrict__ Wc1,
                      const float* __restrict__ bc1, float* __restrict__ out, int G) {
  int i = blockIdx.x * blockDim.x + threadIdx.x;
  if (i >= G * 64) return;
  int g = i >> 6, j = i & 63;
  const float* p = pooled + (size_t)g * 128;
  float s = bc1[j];
  for (int k = 0; k < 128; ++k) s += p[k] * Wc1[k * 64 + j];
  out[i] = fmaxf(s, 0.0f);
}

__global__ void k_fc2(const float* __restrict__ f1, const float* __restrict__ Wc2,
                      const float* __restrict__ bc2, float* __restrict__ out, int G) {
  int g = blockIdx.x * blockDim.x + threadIdx.x;
  if (g >= G) return;
  const float* p = f1 + (size_t)g * 64;
  float s = bc2[0];
  for (int k = 0; k < 64; ++k) s += p[k] * Wc2[k];
  out[g] = s;
}

// ---------------- launch ----------------

extern "C" void kernel_launch(void* const* d_in, const int* in_sizes, int n_in,
                              void* d_out, int out_size, void* d_ws, size_t ws_size,
                              hipStream_t stream) {
  const float* x     = (const float*)d_in[0];
  const int*   ei    = (const int*)d_in[1];
  const int*   batch = (const int*)d_in[2];
  const float* Wl1 = (const float*)d_in[3],  *Wr1 = (const float*)d_in[4];
  const float* att1= (const float*)d_in[5],  *b1  = (const float*)d_in[6];
  const float* g1  = (const float*)d_in[7],  *be1 = (const float*)d_in[8];
  const float* Wl2 = (const float*)d_in[9],  *Wr2 = (const float*)d_in[10];
  const float* att2= (const float*)d_in[11], *b2  = (const float*)d_in[12];
  const float* g2  = (const float*)d_in[13], *be2 = (const float*)d_in[14];
  const float* Wl3 = (const float*)d_in[15], *Wr3 = (const float*)d_in[16];
  const float* att3= (const float*)d_in[17], *b3  = (const float*)d_in[18];
  const float* g3  = (const float*)d_in[19], *be3 = (const float*)d_in[20];
  const float* Wc1 = (const float*)d_in[21], *bc1 = (const float*)d_in[22];
  const float* Wc2 = (const float*)d_in[23], *bc2 = (const float*)d_in[24];
  float* out = (float*)d_out;
  (void)n_in; (void)ws_size;

  const int F_IN = 16, C = 64, HID = 256;
  const int N = in_sizes[0] / F_IN;
  const int E = in_sizes[1] / 2;
  const int G = out_size;
  const int EL = E + N;
  const int* srcA = ei;
  const int* dstA = ei + E;
  const float NEG_INF = -__builtin_huge_valf();

  // workspace carve-out (256B aligned)
  uint8_t* ws = (uint8_t*)d_ws;
  size_t off = 0;
  auto alloc = [&](size_t bytes) -> void* {
    void* p = ws + off;
    off = (off + bytes + 255) & ~(size_t)255;
    return p;
  };
  float*    xl   = (float*)   alloc((size_t)N * HID * 4);
  float*    xr   = (float*)   alloc((size_t)N * HID * 4);
  float*    agg  = (float*)   alloc((size_t)N * HID * 4);
  _Float16* Hh   = (_Float16*)alloc((size_t)N * HID * 2);
  _Float16* Xh   = (_Float16*)alloc((size_t)N * 32 * 2);
  float*    h3   = (float*)   alloc((size_t)N * C * 4);
  float*    mbuf = (float*)   alloc((size_t)N * 4 * 4);
  float*    dbuf = (float*)   alloc((size_t)N * 4 * 4);
  float*    ewb  = (float*)   alloc((size_t)EL * 4 * 4);
  _Float16* WTa  = (_Float16*)alloc((size_t)HID * HID * 2);
  _Float16* WTb  = (_Float16*)alloc((size_t)HID * HID * 2);
  float*    bsum = (float*)   alloc((size_t)HID * 4);
  float*    bsq  = (float*)   alloc((size_t)HID * 4);
  float*    gsum = (float*)   alloc((size_t)G * C * 4);
  float*    gmax = (float*)   alloc((size_t)G * C * 4);
  float*    gcnt = (float*)   alloc((size_t)G * 4);
  float*    pooled = (float*) alloc((size_t)G * 2 * C * 4);
  float*    f1   = (float*)   alloc((size_t)G * C * 4);

  const int TB = 256;
  auto nb = [](long long n, int tb) { return (int)((n + tb - 1) / tb); };

  // attention + BN pass shared by all three layers
  auto gat_pass = [&](const float* att, const float* bias, const float* gg, const float* bb,
                      int Hn, int HIDn, int do_elu, float* fout, _Float16* hout) {
    k_fill<<<nb((long long)N * Hn, TB), TB, 0, stream>>>(mbuf, NEG_INF, N * Hn);
    k_fill<<<nb((long long)N * Hn, TB), TB, 0, stream>>>(dbuf, 0.0f, N * Hn);
    k_fill<<<nb((long long)N * HIDn, TB), TB, 0, stream>>>(agg, 0.0f, N * HIDn);
    k_fill<<<1, TB, 0, stream>>>(bsum, 0.0f, HIDn);
    k_fill<<<1, TB, 0, stream>>>(bsq, 0.0f, HIDn);
    k_edge_logits<<<nb((long long)EL * 32, TB), TB, 0, stream>>>(
        xl, xr, att, srcA, dstA, mbuf, ewb, E, N, Hn);
    k_edge_exp<<<nb((long long)EL * Hn, TB), TB, 0, stream>>>(
        dstA, mbuf, ewb, dbuf, E, N, Hn);
    k_edge_aggregate<<<nb((long long)EL * 32, TB), TB, 0, stream>>>(
        xl, ewb, dbuf, srcA, dstA, agg, E, N, Hn);
    const int RPB = 64;
    k_bias_bnstats<<<nb(N, RPB), HIDn, 0, stream>>>(agg, bias, bsum, bsq, N, HIDn, RPB);
    k_bn_apply<<<nb((long long)N * HIDn, TB), TB, 0, stream>>>(
        agg, bsum, bsq, gg, bb, fout, hout, N, HIDn, do_elu);
  };

  // ---- Layer 1: 16 -> 256 (4 heads x 64, concat) ----
  k_convert_pad_half<<<nb((long long)N * 32, TB), TB, 0, stream>>>(x, Xh, N, F_IN, 32);
  k_transpose_half<<<nb((long long)HID * 32, TB), TB, 0, stream>>>(Wl1, WTa, F_IN, HID, 32);
  k_transpose_half<<<nb((long long)HID * 32, TB), TB, 0, stream>>>(Wr1, WTb, F_IN, HID, 32);
  {
    long long waves = (long long)(N / 16) * (HID / 16);
    k_wmma_gemm<<<nb(waves * 32, TB), TB, 0, stream>>>(Xh, WTa, xl, N, HID, 32);
    k_wmma_gemm<<<nb(waves * 32, TB), TB, 0, stream>>>(Xh, WTb, xr, N, HID, 32);
  }
  gat_pass(att1, b1, g1, be1, 4, 256, /*elu*/1, nullptr, Hh);

  // ---- Layer 2: 256 -> 256 (4 heads x 64, concat) ----
  k_transpose_half<<<nb((long long)HID * HID, TB), TB, 0, stream>>>(Wl2, WTa, HID, HID, HID);
  k_transpose_half<<<nb((long long)HID * HID, TB), TB, 0, stream>>>(Wr2, WTb, HID, HID, HID);
  {
    long long waves = (long long)(N / 16) * (HID / 16);
    k_wmma_gemm<<<nb(waves * 32, TB), TB, 0, stream>>>(Hh, WTa, xl, N, HID, HID);
    k_wmma_gemm<<<nb(waves * 32, TB), TB, 0, stream>>>(Hh, WTb, xr, N, HID, HID);
  }
  gat_pass(att2, b2, g2, be2, 4, 256, /*elu*/1, nullptr, Hh);

  // ---- Layer 3: 256 -> 64 (1 head, no elu) ----
  k_transpose_half<<<nb((long long)C * HID, TB), TB, 0, stream>>>(Wl3, WTa, HID, C, HID);
  k_transpose_half<<<nb((long long)C * HID, TB), TB, 0, stream>>>(Wr3, WTb, HID, C, HID);
  {
    long long waves = (long long)(N / 16) * (C / 16);
    k_wmma_gemm<<<nb(waves * 32, TB), TB, 0, stream>>>(Hh, WTa, xl, N, C, HID);
    k_wmma_gemm<<<nb(waves * 32, TB), TB, 0, stream>>>(Hh, WTb, xr, N, C, HID);
  }
  gat_pass(att3, b3, g3, be3, 1, 64, /*elu*/0, h3, nullptr);

  // ---- pooling (mean | max per graph) + classifier ----
  k_fill<<<nb((long long)G * C, TB), TB, 0, stream>>>(gsum, 0.0f, G * C);
  k_fill<<<nb((long long)G * C, TB), TB, 0, stream>>>(gmax, NEG_INF, G * C);
  k_fill<<<nb((long long)G, TB), TB, 0, stream>>>(gcnt, 0.0f, G);
  k_pool_scatter<<<nb((long long)N * C, TB), TB, 0, stream>>>(h3, batch, gsum, gmax, gcnt, N);
  k_pool_finalize<<<nb((long long)G * 2 * C, TB), TB, 0, stream>>>(gsum, gmax, gcnt, pooled, G);
  k_fc1<<<nb((long long)G * C, TB), TB, 0, stream>>>(pooled, Wc1, bc1, f1, G);
  k_fc2<<<nb((long long)G, TB), TB, 0, stream>>>(f1, Wc2, bc2, out, G);
}
